// SAND_20547123544842
// MI455X (gfx1250) — compile-verified, matrix operand order checked by reading, not run on hardware
//
#include <hip/hip_runtime.h>
#include <hip/hip_bf16.h>
#include <math.h>

typedef _Float16 f16;
typedef __attribute__((ext_vector_type(16))) _Float16 v16h;
typedef __attribute__((ext_vector_type(8)))  _Float16 v8h;
typedef __attribute__((ext_vector_type(8)))  float    v8f;

#define BS  32
#define SL  512
#define CL  20
#define DM  256
#define NH  4
#define HD  64

// ---------------------------------------------------------------------------
// WMMA fragment loader: row-major source, 16 rows starting at `base`, K-block
// of 32 halves.  Matches CDNA5 16-bit A/B layout:
//   lanes 0-15:  row = lane,     K halves {0..7, 16..23}
//   lanes 16-31: row = lane-16,  K halves {8..15, 24..31}
// Two 16-byte loads per lane (global or LDS via generic pointer).
// ---------------------------------------------------------------------------
__device__ __forceinline__ v16h load_frag(const f16* __restrict__ base, int ld, int lane) {
    int r    = lane & 15;
    int koff = (lane >> 4) << 3;            // 0 or 8
    const f16* p = base + r * ld + koff;
    v8h lo = *(const v8h*)(p);              // K = koff .. koff+7
    v8h hi = *(const v8h*)(p + 16);         // K = koff+16 .. koff+23
    v16h f;
#pragma unroll
    for (int i = 0; i < 8; ++i) { f[i] = lo[i]; f[8 + i] = hi[i]; }
    return f;
}

__device__ __forceinline__ v8f wmma16(v16h a, v16h b, v8f c) {
    return __builtin_amdgcn_wmma_f32_16x16x32_f16(false, a, false, b,
                                                  (short)0, c, false, false);
}

// ---------------------------------------------------------------------------
// CDNA5 async global->LDS DMA (ASYNCcnt-tracked), via inline asm so it is
// portable across ROCm-7.2 / amdgpu-toolchain builtin arity differences.
// Copies `rows` rows of `row_bytes` (multiple of 16) each; lane-strided 16B
// chunks.  Generic LDS pointers carry the LDS byte offset in addr[31:0].
// ---------------------------------------------------------------------------
__device__ __forceinline__ void async_rows(const f16* g, int gstride_h,
                                           f16* l, int lstride_h,
                                           int rows, int row_bytes, int lane) {
    int cpr    = row_bytes >> 4;                 // 16B chunks per row
    int chunks = rows * cpr;
    unsigned lbase = (unsigned)(unsigned long long)l;
    for (int c = lane; c < chunks; c += 32) {
        int r = c / cpr;
        int w = (c - r * cpr) << 4;
        unsigned     loff = lbase + (unsigned)(r * lstride_h * 2 + w);
        const char*  gp   = (const char*)g + (size_t)r * gstride_h * 2 + w;
        asm volatile("global_load_async_to_lds_b128 %0, %1, off"
                     :: "v"(loff), "v"(gp) : "memory");
    }
}
__device__ __forceinline__ void wait_async_0() { asm volatile("s_wait_asynccnt 0x0" ::: "memory"); }
__device__ __forceinline__ void wait_async_4() { asm volatile("s_wait_asynccnt 0x4" ::: "memory"); }
__device__ __forceinline__ void wait_async_8() { asm volatile("s_wait_asynccnt 0x8" ::: "memory"); }

// ---------------------------------------------------------------------------
// 1) visit embedding sum + bias + positional encoding
// ---------------------------------------------------------------------------
__global__ __launch_bounds__(256)
void k_embed(const int* __restrict__ seqs, const int* __restrict__ lengths,
             const float* __restrict__ emb, const float* __restrict__ bias,
             const float* __restrict__ pe,
             float* __restrict__ xres, f16* __restrict__ x16) {
    int row = blockIdx.x;                  // b*SL + s
    int t   = threadIdx.x;                 // 0..255 (d index)
    int b = row >> 9, s = row & 511;
    const int* codes = seqs + row * CL;
    float sum = 0.f;
#pragma unroll 4
    for (int c = 0; c < CL; ++c) {
        int idx = codes[c];
        sum += emb[(size_t)idx * DM + t];
    }
    sum += bias[t];
    int len = lengths[b];
    int pos = (s < len) ? (s + 1) : 0;
    sum += pe[(size_t)pos * DM + t];
    xres[(size_t)row * DM + t] = sum;
    x16 [(size_t)row * DM + t] = (f16)sum;
}

// ---------------------------------------------------------------------------
// 2) f32 -> f16 weight conversion
// ---------------------------------------------------------------------------
__global__ void k_cvt(const float* __restrict__ src, f16* __restrict__ dst, int n) {
    int i = blockIdx.x * blockDim.x + threadIdx.x;
    if (i < n) dst[i] = (f16)src[i];
}

// ---------------------------------------------------------------------------
// 3) GEMM  C[M,256] = A[M,256] * W[256,256]^T   (A row-major MxK, W row-major NxK)
//    MODE 0: store f16 row-major     (q, k)
//    MODE 1: store f16 scattered as per-head-transposed V:  vt[(b,h,hd),s]
//    MODE 2: store f32 row-major     (fc output)
//    One wave computes a 16x64 tile; blockDim = 256 (8 waves).
// ---------------------------------------------------------------------------
template <int MODE>
__global__ __launch_bounds__(256)
void k_gemm(const f16* __restrict__ A, const f16* __restrict__ W,
            void* __restrict__ out, int Mtiles) {
    int wave = threadIdx.x >> 5;
    int lane = threadIdx.x & 31;
    int wj   = blockIdx.x * 8 + wave;
    int mt   = wj >> 2;
    int n0   = (wj & 3) * 64;
    if (mt >= Mtiles) return;
    const f16* Abase = A + (size_t)mt * 16 * DM;
    v8f acc[4] = {};
    for (int k0 = 0; k0 < DM; k0 += 32) {
        v16h af = load_frag(Abase + k0, DM, lane);
        if (k0 + 32 < DM)   // prefetch the next A K-block (global_prefetch_b8)
            __builtin_prefetch(Abase + (lane & 15) * DM + k0 + 32, 0, 3);
#pragma unroll
        for (int nt = 0; nt < 4; ++nt) {
            v16h bf = load_frag(W + (size_t)(n0 + nt * 16) * DM + k0, DM, lane);
            acc[nt] = wmma16(af, bf, acc[nt]);
        }
    }
    int nl = lane & 15;
    int mb = (lane >> 4) * 8;
#pragma unroll
    for (int nt = 0; nt < 4; ++nt) {
#pragma unroll
        for (int r = 0; r < 8; ++r) {
            int m = mt * 16 + mb + r;
            int n = n0 + nt * 16 + nl;
            float v = acc[nt][r];
            if (MODE == 0) {
                ((f16*)out)[(size_t)m * DM + n] = (f16)v;
            } else if (MODE == 1) {
                // flat index into [bs, sl, D]; reinterpret through the
                // reshape(bs,H,sl,hd) view and store V transposed per head.
                int F  = m * DM + n;
                int b  = F >> 17;            // / (SL*DM)
                int rr = F & (SL * DM - 1);
                int h  = rr >> 15;           // / (SL*HD)
                int r2 = rr & (SL * HD - 1);
                int s  = r2 >> 6;            // / HD
                int hd = r2 & (HD - 1);
                ((f16*)out)[(((b * NH + h) * HD + hd) << 9) + s] = (f16)v;
            } else {
                ((float*)out)[(size_t)m * DM + n] = v;
            }
        }
    }
}

// ---------------------------------------------------------------------------
// 4) attention: 1 wave per (b, h, 16-row tile).  K tiles and V^T k-blocks are
//    double-buffered in LDS via async global->LDS DMA; the 16x512 energy tile
//    stays in LDS (never touches HBM); softmax uses all 32 lanes.
// ---------------------------------------------------------------------------
__global__ __launch_bounds__(32)
void k_attn(const f16* __restrict__ q16, const f16* __restrict__ k16,
            const f16* __restrict__ vt16, const int* __restrict__ masks,
            f16* __restrict__ o16) {
    __shared__ __align__(16) f16 sK [2][16 * HD];   // 2 x 2KB K tile
    __shared__ __align__(16) f16 sVT[2][HD * 32];   // 2 x 4KB V^T k-block
    __shared__ float eLDS[16 * SL];                 // 32KB energies
    __shared__ f16   aLDS[16 * SL];                 // 16KB attn (f16)
    __shared__ float pmax[32], psum[32];

    int lane = threadIdx.x;
    int id = blockIdx.x;               // b*(NH*32) + h*32 + mt
    int mt = id & 31;
    int h  = (id >> 5) & (NH - 1);
    int b  = id >> 7;
    // contiguous per-(b,h) slices produced by the reshape-view
    const f16* Q  = q16  + (size_t)b * SL * DM + (size_t)h * SL * HD;   // [SL,64]
    const f16* K  = k16  + (size_t)b * SL * DM + (size_t)h * SL * HD;   // [SL,64]
    const f16* VT = vt16 + (size_t)(b * NH + h) * HD * SL;              // [64,SL]

    v16h qa0 = load_frag(Q + mt * 16 * HD +  0, HD, lane);
    v16h qa1 = load_frag(Q + mt * 16 * HD + 32, HD, lane);
    int nl = lane & 15;
    int mb = (lane >> 4) * 8;
    const int* mrow = masks + ((size_t)b * SL + mt * 16) * SL;

    // ---- energy = Q K^T / 8, masked; K tiles streamed via async DMA --------
    async_rows(K, HD, &sK[0][0], HD, 16, HD * 2, lane);          // 4 ops/lane
    for (int nt = 0; nt < 32; ++nt) {
        if (nt + 1 < 32) {
            async_rows(K + (nt + 1) * 16 * HD, HD, &sK[(nt + 1) & 1][0],
                       HD, 16, HD * 2, lane);
            wait_async_4();          // in-order completion: current tile done
        } else {
            wait_async_0();
        }
        const f16* kt = &sK[nt & 1][0];
        v8f acc = {};
        v16h kb0 = load_frag(kt +  0, HD, lane);
        v16h kb1 = load_frag(kt + 32, HD, lane);
        acc = wmma16(qa0, kb0, acc);
        acc = wmma16(qa1, kb1, acc);
#pragma unroll
        for (int r = 0; r < 8; ++r) {
            int m = mb + r;
            int n = nt * 16 + nl;
            float e = acc[r] * 0.125f;                 // 1/sqrt(64)
            if (mrow[(size_t)m * SL + n] == 0) e = -__builtin_inff();
            eLDS[m * SL + n] = e;
        }
    }
    __syncthreads();

    // ---- softmax: 2 lanes per row, half a row (256) each -------------------
    {
        int srow = lane & 15, shalf = lane >> 4;
        const float* er = eLDS + srow * SL + shalf * 256;
        float mx = -__builtin_inff();
        for (int n = 0; n < 256; ++n) mx = fmaxf(mx, er[n]);
        pmax[lane] = mx;
        __syncthreads();
        float rmax = fmaxf(pmax[srow], pmax[srow + 16]);
        float ss = 0.f;
        for (int n = 0; n < 256; ++n) ss += __expf(er[n] - rmax);
        psum[lane] = ss;
        __syncthreads();
        float inv = 1.f / (psum[srow] + psum[srow + 16]);
        f16* ar = aLDS + srow * SL + shalf * 256;
        for (int n = 0; n < 256; ++n) ar[n] = (f16)(__expf(er[n] - rmax) * inv);
    }
    __syncthreads();

    // ---- O = attn @ V^T; V^T k-blocks streamed via async DMA ---------------
    v8f oacc[4] = {};
    async_rows(VT, SL, &sVT[0][0], 32, HD, 64, lane);            // 8 ops/lane
    for (int kt = 0; kt < 16; ++kt) {
        if (kt + 1 < 16) {
            async_rows(VT + (kt + 1) * 32, SL, &sVT[(kt + 1) & 1][0],
                       32, HD, 64, lane);
            wait_async_8();
        } else {
            wait_async_0();
        }
        v16h af = load_frag((const f16*)aLDS + kt * 32, SL, lane);
        const f16* vtile = &sVT[kt & 1][0];
#pragma unroll
        for (int nt = 0; nt < 4; ++nt) {
            v16h bf = load_frag(vtile + nt * 16 * 32, 32, lane);
            oacc[nt] = wmma16(af, bf, oacc[nt]);
        }
    }
#pragma unroll
    for (int nt = 0; nt < 4; ++nt) {
#pragma unroll
        for (int r = 0; r < 8; ++r) {
            int s = mt * 16 + mb + r;
            int n = nt * 16 + nl;
            // o.transpose(0,2,1,3).reshape(bs,sl,D)
            o16[((size_t)(b * SL + s)) * DM + h * HD + n] = (f16)oacc[nt][r];
        }
    }
}

// ---------------------------------------------------------------------------
// 5) residual add + LayerNorm + length masking
// ---------------------------------------------------------------------------
__global__ __launch_bounds__(256)
void k_ln(const float* __restrict__ y, const float* __restrict__ xres,
          const int* __restrict__ lengths, const float* __restrict__ g,
          const float* __restrict__ bt, float* __restrict__ xln) {
    __shared__ float red[256];
    __shared__ float red2[256];
    int row = blockIdx.x;
    int t   = threadIdx.x;
    float v = y[(size_t)row * DM + t] + xres[(size_t)row * DM + t];
    red[t] = v;
    __syncthreads();
    for (int s2 = 128; s2 > 0; s2 >>= 1) {
        if (t < s2) red[t] += red[t + s2];
        __syncthreads();
    }
    float mu = red[0] * (1.0f / DM);
    __syncthreads();
    float dv = v - mu;
    red2[t] = dv * dv;
    __syncthreads();
    for (int s2 = 128; s2 > 0; s2 >>= 1) {
        if (t < s2) red2[t] += red2[t + s2];
        __syncthreads();
    }
    float var = red2[0] * (1.0f / DM);
    float out = dv * rsqrtf(var + 1e-5f) * g[t] + bt[t];
    int b = row >> 9, s = row & 511;
    if (s >= lengths[b]) out = 0.f;
    xln[(size_t)row * DM + t] = out;
}

// ---------------------------------------------------------------------------
// 6) triangular-weighted temporal pooling into 4 bins
// ---------------------------------------------------------------------------
__global__ __launch_bounds__(256)
void k_pool(const float* __restrict__ xln, const int* __restrict__ lengths,
            float* __restrict__ U) {
    int id = blockIdx.x;                 // b*4 + mbin
    int b = id >> 2, mbin = id & 3;
    int t = threadIdx.x;
    float len = (float)lengths[b];
    float mfl = (float)(mbin + 1);
    float acc = 0.f;
    for (int s = 0; s < SL; ++s) {
        float sv = 4.0f * (float)(s + 1) / len;
        float w  = 1.0f - fabsf(sv - mfl) * 0.25f;
        acc += w * w * xln[((size_t)(b * SL + s)) * DM + t];
    }
    U[(size_t)id * DM + t] = acc;
}

// ---------------------------------------------------------------------------
// 7) final projection [32,1024] @ [1024,2]^T + bias
// ---------------------------------------------------------------------------
__global__ __launch_bounds__(256)
void k_out(const float* __restrict__ U, const float* __restrict__ ow,
           const float* __restrict__ ob, float* __restrict__ out) {
    __shared__ float r0[256], r1[256];
    int b = blockIdx.x, t = threadIdx.x;
    float a0 = 0.f, a1 = 0.f;
    for (int i = t; i < 4 * DM; i += 256) {
        float u = U[(size_t)b * 4 * DM + i];
        a0 += u * ow[i];
        a1 += u * ow[4 * DM + i];
    }
    r0[t] = a0; r1[t] = a1;
    __syncthreads();
    for (int s2 = 128; s2 > 0; s2 >>= 1) {
        if (t < s2) { r0[t] += r0[t + s2]; r1[t] += r1[t + s2]; }
        __syncthreads();
    }
    if (t == 0) {
        out[b * 2 + 0] = r0[0] + ob[0];
        out[b * 2 + 1] = r1[0] + ob[1];
    }
}

// ---------------------------------------------------------------------------
extern "C" void kernel_launch(void* const* d_in, const int* in_sizes, int n_in,
                              void* d_out, int out_size, void* d_ws, size_t ws_size,
                              hipStream_t stream) {
    (void)in_sizes; (void)n_in; (void)out_size; (void)ws_size;
    const int*   seqs    = (const int*)  d_in[0];
    const int*   masks   = (const int*)  d_in[1];
    const int*   lengths = (const int*)  d_in[2];
    // d_in[3] seq_time_step, d_in[4] code_masks: unused by the reference
    const float* emb     = (const float*)d_in[5];
    const float* bias    = (const float*)d_in[6];
    const float* pe      = (const float*)d_in[7];
    const float* Wq      = (const float*)d_in[8];
    const float* Wk      = (const float*)d_in[9];
    const float* Wv      = (const float*)d_in[10];
    const float* Wfc     = (const float*)d_in[11];
    const float* ln_g    = (const float*)d_in[12];
    const float* ln_b    = (const float*)d_in[13];
    const float* out_w   = (const float*)d_in[14];
    const float* out_b   = (const float*)d_in[15];
    float* out = (float*)d_out;

    const size_t ROWS = (size_t)BS * SL;        // 16384
    size_t off = 0;
    auto alloc = [&](size_t bytes) {
        void* p = (char*)d_ws + off;
        off += (bytes + 255) & ~(size_t)255;
        return p;
    };
    float* xres  = (float*)alloc(ROWS * DM * sizeof(float));   // 16 MB
    f16*   x16   = (f16*)  alloc(ROWS * DM * sizeof(f16));     //  8 MB
    f16*   q16   = (f16*)  alloc(ROWS * DM * sizeof(f16));
    f16*   k16   = (f16*)  alloc(ROWS * DM * sizeof(f16));
    f16*   vt16  = (f16*)  alloc(ROWS * DM * sizeof(f16));
    f16*   o16   = (f16*)  alloc(ROWS * DM * sizeof(f16));
    float* y32   = (float*)alloc(ROWS * DM * sizeof(float));
    float* xln   = (float*)alloc(ROWS * DM * sizeof(float));
    f16*   wq16  = (f16*)  alloc((size_t)DM * DM * sizeof(f16));
    f16*   wk16  = (f16*)  alloc((size_t)DM * DM * sizeof(f16));
    f16*   wv16  = (f16*)  alloc((size_t)DM * DM * sizeof(f16));
    f16*   wfc16 = (f16*)  alloc((size_t)DM * DM * sizeof(f16));
    float* U     = (float*)alloc((size_t)BS * 4 * DM * sizeof(float));

    const int WN = DM * DM;                     // 65536
    k_cvt<<<WN / 256, 256, 0, stream>>>(Wq,  wq16,  WN);
    k_cvt<<<WN / 256, 256, 0, stream>>>(Wk,  wk16,  WN);
    k_cvt<<<WN / 256, 256, 0, stream>>>(Wv,  wv16,  WN);
    k_cvt<<<WN / 256, 256, 0, stream>>>(Wfc, wfc16, WN);

    k_embed<<<(int)ROWS, 256, 0, stream>>>(seqs, lengths, emb, bias, pe, xres, x16);

    const int Mtiles = (int)(ROWS / 16);        // 1024
    const int gblk   = Mtiles * 4 / 8;          // 512 blocks of 8 waves
    k_gemm<0><<<gblk, 256, 0, stream>>>(x16, wq16, (void*)q16,  Mtiles);
    k_gemm<0><<<gblk, 256, 0, stream>>>(x16, wk16, (void*)k16,  Mtiles);
    k_gemm<1><<<gblk, 256, 0, stream>>>(x16, wv16, (void*)vt16, Mtiles);

    k_attn<<<BS * NH * (SL / 16), 32, 0, stream>>>(q16, k16, vt16, masks, o16);

    k_gemm<2><<<gblk, 256, 0, stream>>>(o16, wfc16, (void*)y32, Mtiles);

    k_ln<<<(int)ROWS, 256, 0, stream>>>(y32, xres, lengths, ln_g, ln_b, xln);
    k_pool<<<BS * 4, 256, 0, stream>>>(xln, lengths, U);
    k_out<<<BS, 256, 0, stream>>>(U, out_w, out_b, out);
}
